// BiasedSelfAttentionLayer_59803124630031
// MI455X (gfx1250) — matrix operand order repaired and checked by reading, not compiled, not run on hardware
//
#include <hip/hip_runtime.h>
#include <hip/hip_bf16.h>
#include <math.h>
#include <stdint.h>

typedef __attribute__((ext_vector_type(16))) _Float16 v16h;
typedef __attribute__((ext_vector_type(8)))  _Float16 v8h;
typedef __attribute__((ext_vector_type(4)))  _Float16 v4h;
typedef __attribute__((ext_vector_type(8)))  float    v8f;
typedef __attribute__((ext_vector_type(4)))  unsigned int u32x4;
typedef __attribute__((ext_vector_type(8)))  int i32x8;
typedef __attribute__((ext_vector_type(4)))  int i32x4;

#define Bc 64
#define Nc 256
#define Hc 16
#define Dc 1024
#define DHc 64
#define Mrows (Bc * Nc)       // 16384
#define SCALEF 0.125f         // DH^-0.5
#define EPSF 1e-5f

#if __has_builtin(__builtin_amdgcn_tensor_load_to_lds)
#define HAVE_TDM 1
#else
#define HAVE_TDM 0
#endif

static __device__ inline v16h pack16(v8h lo, v8h hi) {
    v16h r;
#pragma unroll
    for (int i = 0; i < 8; ++i) { r[i] = lo[i]; r[i + 8] = hi[i]; }
    return r;
}

static __device__ inline v8f wmma_f16(v16h a, v16h b, v8f c) {
    return __builtin_amdgcn_wmma_f32_16x16x32_f16(false, a, false, b, (short)0, c, false, false);
}

#if HAVE_TDM
// TDM: load a 64(rows) x 32(K halves) f16 tile from a row-major [*,1024] tensor
// into LDS (row-major [64][32]).  D# built per CDNA5 ISA.
// 6-arg builtin form (amdgpu-toolchain / clang-23):
//   (u32x4 g0, i32x8 g1, i32x4 g2, i32x4 g3, i32x8 pad, i32 cpol)
static __device__ inline void tdm_load_w_tile(const _Float16* gsrc, unsigned ldsByte) {
    unsigned long long ga = (unsigned long long)(uintptr_t)gsrc;
    u32x4 g0;
    g0[0] = 1u;                                                 // count=1, user D#
    g0[1] = ldsByte;                                            // lds_addr
    g0[2] = (unsigned)ga;                                       // global_addr[31:0]
    g0[3] = ((unsigned)(ga >> 32) & 0x01FFFFFFu) | (2u << 30);  // global_addr[56:32] | type=2
    i32x8 g1;
    g1[0] = 1 << 16;             // workgroup_mask=0, data_size=1 (2 bytes)
    g1[1] = (int)(1024u << 16);  // atomic_barrier=0 | tensor_dim0[15:0]=1024
    g1[2] = (int)(1024u << 16);  // tensor_dim0[31:16]=0 | tensor_dim1[15:0]=1024
    g1[3] = (int)(32u << 16);    // tensor_dim1[31:16]=0 | tile_dim0=32
    g1[4] = 64;                  // tile_dim1=64 | tile_dim2=0
    g1[5] = 1024;                // tensor_dim0_stride[31:0]=1024
    g1[6] = 0;                   // stride hi | tensor_dim1_stride lo
    g1[7] = 0;
    i32x4 z; z[0] = 0; z[1] = 0; z[2] = 0; z[3] = 0;
    i32x8 z8;
#pragma unroll
    for (int i = 0; i < 8; ++i) z8[i] = 0;
    __builtin_amdgcn_tensor_load_to_lds(g0, g1, z, z, z8, 0);
}
#endif

// ---------------------------------------------------------------- offsets ---
__global__ void offsets_kernel(const int* __restrict__ batch, int T,
                               int* __restrict__ off, int* __restrict__ cnt) {
    int b = threadIdx.x;
    if (b >= Bc) return;
    int lo = 0, hi = T;
    while (lo < hi) { int mid = (lo + hi) >> 1; if (batch[mid] < b) lo = mid + 1; else hi = mid; }
    int lo2 = 0, hi2 = T;
    while (lo2 < hi2) { int mid = (lo2 + hi2) >> 1; if (batch[mid] < b + 1) lo2 = mid + 1; else hi2 = mid; }
    off[b] = lo;
    cnt[b] = lo2 - lo;
}

// -------------------------------------------------------- f32 -> f16 convert
__global__ __launch_bounds__(256) void cvt_f16_kernel(const float* __restrict__ src,
                                                      _Float16* __restrict__ dst) {
    int e = (blockIdx.x * 256 + threadIdx.x) * 4;
    float4 v = *(const float4*)(src + e);
    v4h o;
    o[0] = (_Float16)v.x; o[1] = (_Float16)v.y; o[2] = (_Float16)v.z; o[3] = (_Float16)v.w;
    *(v4h*)(dst + e) = o;
}

// ---------------------------------------------------------------- densify ---
__global__ __launch_bounds__(256) void densify_kernel(const float* __restrict__ x,
                                                      const int* __restrict__ off,
                                                      const int* __restrict__ cnt,
                                                      _Float16* __restrict__ xd) {
    long e = ((long)blockIdx.x * 256 + threadIdx.x) * 4;
    int b = (int)(e / ((long)Nc * Dc));
    long rem = e % ((long)Nc * Dc);
    int j = (int)(rem / Dc);
    int d = (int)(rem % Dc);
    v4h o;
    if (j < cnt[b]) {
        long t = (long)off[b] + j;
        float4 v = *(const float4*)(x + t * Dc + d);
        o[0] = (_Float16)v.x; o[1] = (_Float16)v.y; o[2] = (_Float16)v.z; o[3] = (_Float16)v.w;
    } else {
        o[0] = (_Float16)0.f; o[1] = (_Float16)0.f; o[2] = (_Float16)0.f; o[3] = (_Float16)0.f;
    }
    *(v4h*)(xd + e) = o;
}

// ------------------------------------------------------------- WMMA GEMM ----
// C[m,n] = sum_k A[m,k] * Wt[n,k] + bias[n]
// mode 0: f16 row-major; mode 1: f16 V-transposed (B,H,DH,N); mode 2: f32 row-major
__global__ __launch_bounds__(256) void gemm_wmma_kernel(const _Float16* __restrict__ A,
                                                        const _Float16* __restrict__ Wt,
                                                        const float* __restrict__ bias,
                                                        void* __restrict__ out, int mode) {
    __shared__ _Float16 ldsB[2][64 * 32];   // double-buffered 64 cols x 32-K tile (2x4KB)
    const int tid  = threadIdx.x;
    const int lane = tid & 31;
    const int wave = tid >> 5;
    const int l15  = lane & 15;
    const int lh   = lane >> 4;
    const int kb   = lh * 8;
    const int m0   = (blockIdx.x * 8 + wave) * 16;
    const int colbase = blockIdx.y * 64;

    v8f acc[4];
#pragma unroll
    for (int nt = 0; nt < 4; ++nt)
#pragma unroll
        for (int i = 0; i < 8; ++i) acc[nt][i] = 0.f;

    const _Float16* arow  = A + (long)(m0 + l15) * Dc;
    const _Float16* wbase = Wt + (long)colbase * Dc;

#if HAVE_TDM
    unsigned ldsAddr0 = (unsigned)(uintptr_t)&ldsB[0][0];
    unsigned ldsAddr1 = (unsigned)(uintptr_t)&ldsB[1][0];
    if (wave == 0) {
        tdm_load_w_tile(wbase, ldsAddr0);
        __builtin_amdgcn_s_wait_tensorcnt(0);
    }
    __syncthreads();
#else
    const int snl = tid >> 2;            // 0..63
    const int sko = (tid & 3) * 8;       // 0,8,16,24
    *(v8h*)&ldsB[0][snl * 32 + sko] = *(const v8h*)&wbase[(long)snl * Dc + sko];
    __syncthreads();
#endif

    v8h a0 = *(const v8h*)(arow + kb);
    v8h a1 = *(const v8h*)(arow + 16 + kb);

    for (int kc = 0; kc < 32; ++kc) {
        const int k0  = kc * 32;
        const int cur = kc & 1;

        // ---- kick off staging of next K-chunk before computing current ----
#if HAVE_TDM
        if ((kc + 1 < 32) && (wave == 0))
            tdm_load_w_tile(wbase + (k0 + 32), (kc & 1) ? ldsAddr0 : ldsAddr1);
#else
        v8h st = a0;  // dummy init
        if (kc + 1 < 32)
            st = *(const v8h*)&wbase[(long)snl * Dc + (k0 + 32) + sko];
#endif
        v8h n0 = a0, n1 = a1;
        if (kc + 1 < 32) {
            n0 = *(const v8h*)(arow + (k0 + 32) + kb);
            n1 = *(const v8h*)(arow + (k0 + 48) + kb);
        }

        // ---- compute current chunk from LDS ----
        v16h a = pack16(a0, a1);
        const _Float16* bufp = &ldsB[cur][0];
#pragma unroll
        for (int nt = 0; nt < 4; ++nt) {
            const _Float16* bp = bufp + (nt * 16 + l15) * 32 + kb;
            v16h bf = pack16(*(const v8h*)bp, *(const v8h*)(bp + 16));
            acc[nt] = wmma_f16(a, bf, acc[nt]);
        }
        a0 = n0; a1 = n1;

        // ---- publish next buffer ----
#if HAVE_TDM
        if (wave == 0) __builtin_amdgcn_s_wait_tensorcnt(0);
        __syncthreads();
#else
        if (kc + 1 < 32)
            *(v8h*)&ldsB[cur ^ 1][snl * 32 + sko] = st;
        __syncthreads();
#endif
    }

#pragma unroll
    for (int nt = 0; nt < 4; ++nt) {
#pragma unroll
        for (int i = 0; i < 8; ++i) {
            int m = m0 + i + 8 * lh;
            int n = colbase + nt * 16 + l15;
            float v = acc[nt][i] + bias[n];
            if (mode == 0) {
                ((_Float16*)out)[(long)m * Dc + n] = (_Float16)v;
            } else if (mode == 1) {
                int bb = m >> 8, j = m & 255;
                int hh = n >> 6, dh = n & 63;
                ((_Float16*)out)[(((long)bb * Hc + hh) * DHc + dh) * Nc + j] = (_Float16)v;
            } else {
                ((float*)out)[(long)m * Dc + n] = v;
            }
        }
    }
}

// ------------------------------------------------------------- attention ----
__global__ __launch_bounds__(256) void attn_kernel(const _Float16* __restrict__ qf,
                                                   const _Float16* __restrict__ kf,
                                                   const _Float16* __restrict__ vT,
                                                   const float* __restrict__ bias,
                                                   const int* __restrict__ cnt,
                                                   _Float16* __restrict__ attnout) {
    __shared__ _Float16 ldsP[8 * 16 * 256];   // 64 KB: per-wave 16x256 P tile
    const int b = blockIdx.z, h = blockIdx.y;
    const int lane = threadIdx.x & 31;
    const int wave = threadIdx.x >> 5;
    const int l15  = lane & 15;
    const int lh   = lane >> 4;
    const int kb   = lh * 8;
    const int m0   = blockIdx.x * 128 + wave * 16;
    const int count = cnt[b];

    // ---- S = Q K^T ----
    const _Float16* qrow = qf + (long)(b * Nc + m0 + l15) * Dc + h * DHc;
    v16h aq0 = pack16(*(const v8h*)(qrow + kb),      *(const v8h*)(qrow + 16 + kb));
    v16h aq1 = pack16(*(const v8h*)(qrow + 32 + kb), *(const v8h*)(qrow + 48 + kb));

    v8f s[16];
#pragma unroll
    for (int t = 0; t < 16; ++t)
#pragma unroll
        for (int i = 0; i < 8; ++i) s[t][i] = 0.f;

#pragma unroll
    for (int kt = 0; kt < 16; ++kt) {
        const _Float16* krow = kf + (long)(b * Nc + kt * 16 + l15) * Dc + h * DHc;
        v16h b0 = pack16(*(const v8h*)(krow + kb),      *(const v8h*)(krow + 16 + kb));
        v16h b1 = pack16(*(const v8h*)(krow + 32 + kb), *(const v8h*)(krow + 48 + kb));
        s[kt] = wmma_f16(aq0, b0, s[kt]);
        s[kt] = wmma_f16(aq1, b1, s[kt]);
    }

    // ---- scale + bias + column mask ----
    const float* brow = bias + (((long)b * Hc + h) * Nc) * Nc;
#pragma unroll
    for (int kt = 0; kt < 16; ++kt) {
        int n = kt * 16 + l15;
        bool valid = (n < count);
#pragma unroll
        for (int i = 0; i < 8; ++i) {
            int m = m0 + i + 8 * lh;
            float v = s[kt][i] * SCALEF + brow[(long)m * Nc + n];
            s[kt][i] = valid ? v : -1e30f;
        }
    }

    // ---- softmax over n ----
    _Float16* pw = &ldsP[wave * 16 * 256];
#pragma unroll
    for (int i = 0; i < 8; ++i) {
        float mx = -1e30f;
#pragma unroll
        for (int kt = 0; kt < 16; ++kt) mx = fmaxf(mx, s[kt][i]);
#pragma unroll
        for (int o = 1; o < 16; o <<= 1) mx = fmaxf(mx, __shfl_xor(mx, o, 32));
        float sum = 0.f;
#pragma unroll
        for (int kt = 0; kt < 16; ++kt) { float e = __expf(s[kt][i] - mx); s[kt][i] = e; sum += e; }
#pragma unroll
        for (int o = 1; o < 16; o <<= 1) sum += __shfl_xor(sum, o, 32);
        float inv = 1.f / sum;
        int ml = i + 8 * lh;
#pragma unroll
        for (int kt = 0; kt < 16; ++kt)
            pw[ml * 256 + kt * 16 + l15] = (_Float16)(s[kt][i] * inv);
    }
    __syncthreads();

    // ---- O = P V  (V pre-transposed: (B,H,DH,N)) ----
    v8f o[4];
#pragma unroll
    for (int nt = 0; nt < 4; ++nt)
#pragma unroll
        for (int i = 0; i < 8; ++i) o[nt][i] = 0.f;

    const _Float16* vbase = vT + ((long)b * Hc + h) * DHc * Nc;
#pragma unroll
    for (int kt2 = 0; kt2 < 8; ++kt2) {
        const _Float16* prow = pw + (long)l15 * 256 + kt2 * 32;
        v16h ap = pack16(*(const v8h*)(prow + kb), *(const v8h*)(prow + 16 + kb));
#pragma unroll
        for (int nt = 0; nt < 4; ++nt) {
            const _Float16* vp = vbase + (long)(nt * 16 + l15) * Nc + kt2 * 32 + kb;
            v16h bf = pack16(*(const v8h*)vp, *(const v8h*)(vp + 16));
            o[nt] = wmma_f16(ap, bf, o[nt]);
        }
    }

#pragma unroll
    for (int nt = 0; nt < 4; ++nt)
#pragma unroll
        for (int i = 0; i < 8; ++i) {
            int m = m0 + i + 8 * lh;
            int c = h * DHc + nt * 16 + l15;
            attnout[(long)(b * Nc + m) * Dc + c] = (_Float16)o[nt][i];
        }
}

// ---------------------------------------------- gather + residual + LN ------
__global__ __launch_bounds__(256) void resid_ln_kernel(const float* __restrict__ proj,
                                                       const float* __restrict__ x,
                                                       const int* __restrict__ batch,
                                                       const int* __restrict__ off,
                                                       const float* __restrict__ gamma,
                                                       const float* __restrict__ beta,
                                                       float* __restrict__ out) {
    __shared__ float rs[256];
    __shared__ float rq[256];
    int t = blockIdx.x;
    int b = batch[t];
    long r = (long)b * Nc + (t - off[b]);
    int c0 = threadIdx.x * 4;

    float4 pv = *(const float4*)(proj + r * Dc + c0);
    float4 xv = *(const float4*)(x + (long)t * Dc + c0);
    float v0 = pv.x + xv.x, v1 = pv.y + xv.y, v2 = pv.z + xv.z, v3 = pv.w + xv.w;

    float sm = v0 + v1 + v2 + v3;
    float sq = v0 * v0 + v1 * v1 + v2 * v2 + v3 * v3;
    rs[threadIdx.x] = sm; rq[threadIdx.x] = sq;
    __syncthreads();
    for (int st = 128; st > 0; st >>= 1) {
        if (threadIdx.x < st) { rs[threadIdx.x] += rs[threadIdx.x + st]; rq[threadIdx.x] += rq[threadIdx.x + st]; }
        __syncthreads();
    }
    float mu = rs[0] * (1.0f / Dc);
    float var = rq[0] * (1.0f / Dc) - mu * mu;
    float rinv = rsqrtf(var + EPSF);

    float4 g = *(const float4*)(gamma + c0);
    float4 be = *(const float4*)(beta + c0);
    float4 ov;
    ov.x = (v0 - mu) * rinv * g.x + be.x;
    ov.y = (v1 - mu) * rinv * g.y + be.y;
    ov.z = (v2 - mu) * rinv * g.z + be.z;
    ov.w = (v3 - mu) * rinv * g.w + be.w;
    *(float4*)(out + (long)t * Dc + c0) = ov;
}

// ---------------------------------------------------------------------------
extern "C" void kernel_launch(void* const* d_in, const int* in_sizes, int n_in,
                              void* d_out, int out_size, void* d_ws, size_t ws_size,
                              hipStream_t stream) {
    const float* x     = (const float*)d_in[0];
    const int*   batch = (const int*)d_in[1];
    const float* abias = (const float*)d_in[2];
    const float* Wq = (const float*)d_in[3];  const float* bq = (const float*)d_in[4];
    const float* Wk = (const float*)d_in[5];  const float* bk = (const float*)d_in[6];
    const float* Wv = (const float*)d_in[7];  const float* bv = (const float*)d_in[8];
    const float* Wp = (const float*)d_in[9];  const float* bp = (const float*)d_in[10];
    const float* gamma = (const float*)d_in[11];
    const float* beta  = (const float*)d_in[12];
    float* out = (float*)d_out;
    const int T = in_sizes[1];

    char* w = (char*)d_ws;
    int*  off = (int*)w;                     // 64 ints
    int*  cnt = (int*)(w + 512);             // 64 ints
    const long HB = (long)Mrows * Dc * sizeof(_Float16);   // 33,554,432
    _Float16* xd      = (_Float16*)(w + 1024);
    _Float16* qf      = (_Float16*)(w + 1024 + 1 * HB);
    _Float16* kf      = (_Float16*)(w + 1024 + 2 * HB);
    _Float16* vT      = (_Float16*)(w + 1024 + 3 * HB);
    _Float16* attnout = (_Float16*)(w + 1024 + 4 * HB);
    _Float16* Wq16    = (_Float16*)(w + 1024 + 5 * HB);
    _Float16* Wk16    = Wq16 + (long)Dc * Dc;
    _Float16* Wv16    = Wk16 + (long)Dc * Dc;
    _Float16* Wp16    = Wv16 + (long)Dc * Dc;
    float*    proj    = (float*)((char*)(Wp16 + (long)Dc * Dc));

    offsets_kernel<<<1, 64, 0, stream>>>(batch, T, off, cnt);

    const int cvtGrid = (Dc * Dc) / (256 * 4);  // 1024
    cvt_f16_kernel<<<cvtGrid, 256, 0, stream>>>(Wq, Wq16);
    cvt_f16_kernel<<<cvtGrid, 256, 0, stream>>>(Wk, Wk16);
    cvt_f16_kernel<<<cvtGrid, 256, 0, stream>>>(Wv, Wv16);
    cvt_f16_kernel<<<cvtGrid, 256, 0, stream>>>(Wp, Wp16);

    densify_kernel<<<(Bc * Nc * Dc) / (256 * 4), 256, 0, stream>>>(x, off, cnt, xd);

    dim3 ggrid(Mrows / (8 * 16), Dc / 64);  // (128, 16)
    gemm_wmma_kernel<<<ggrid, 256, 0, stream>>>(xd, Wq16, bq, (void*)qf, 0);
    gemm_wmma_kernel<<<ggrid, 256, 0, stream>>>(xd, Wk16, bk, (void*)kf, 0);
    gemm_wmma_kernel<<<ggrid, 256, 0, stream>>>(xd, Wv16, bv, (void*)vT, 1);

    attn_kernel<<<dim3(2, Hc, Bc), 256, 0, stream>>>(qf, kf, vT, abias, cnt, attnout);

    gemm_wmma_kernel<<<ggrid, 256, 0, stream>>>(attnout, Wp16, bp, (void*)proj, 2);

    resid_ln_kernel<<<T, 256, 0, stream>>>(proj, x, batch, off, gamma, beta, out);
}
